// GATLSTMPolicyNetwork_15470472200530
// MI455X (gfx1250) — matrix operand order repaired and checked by reading, not compiled, not run on hardware
//
#include <hip/hip_runtime.h>
#include <hip/hip_bf16.h>
#include <math.h>

// ---------------------------------------------------------------------------
// GAT-LSTM policy net for MI455X (gfx1250, wave32, WMMA f32_16x16x32_f16)
//
// Kernel 1 (sequential, 1 block x 512 thr = 16 waves): LSTM over T=4096.
//   gates[64,256] = [h | x_t | 0][64,96] @ WA[96,256]  (augmented-K WMMA GEMM)
//   - ping-pong A-panel in LDS: step t reads buf[t&1], writes h(t), x(t+1)
//     into buf[(t+1)&1]  -> race-free with ONE barrier per step
//   - wave w (mt=w&3, s=w>>2) owns M-tile mt and N-tiles {s,4+s,8+s,12+s}:
//     its 4 accumulators are i/f/g/o for the SAME (node, hidden) elements,
//     so the cell update is pure per-lane VALU (c held in VGPRs)
//   - weights live in VGPRs as WMMA B-fragments for the whole kernel
//   - node-pool partial sums go straight to global; reduced in kernel 2
//     (keeps the reduction OFF the sequential critical path)
//
// Kernel 2 (parallel, 256 blocks x 128 thr): per 16 timesteps,
//   hbar = mean_n h; g = relu(hbar@W_gat^T+b_gat); mean/std heads via WMMA.
//   (GAT attention == identity: adjacency is eye -> 1-entry softmax == 1,
//    and mean-pool commutes with the linear layer.)
// ---------------------------------------------------------------------------

typedef _Float16 v16h __attribute__((ext_vector_type(16)));
typedef _Float16 v8h  __attribute__((ext_vector_type(8)));
typedef float    v8f  __attribute__((ext_vector_type(8)));

#define T_SEQ 4096
#define OBS 512
#define HA_STRIDE 88    // halves: 64 h | 8 x | 16 pad-zero; 176B = 11*16B
#define HA_BUF (64 * HA_STRIDE)
#define GS_STRIDE 264   // halves: 256 g | 8 pad

__device__ __forceinline__ float fast_sig(float x) {
  return __builtin_amdgcn_rcpf(1.f + __expf(-x));
}

__device__ __forceinline__ float fast_tanh(float x) {
#if __has_builtin(__builtin_amdgcn_tanhf)
  return __builtin_amdgcn_tanhf(x);           // native V_TANH_F32 on gfx1250
#elif __has_builtin(__builtin_amdgcn_tanh_f32)
  return __builtin_amdgcn_tanh_f32(x);
#else
  // tanh(x) = 1 - 2/(exp(2x)+1), branch-free
  const float e = __expf(2.f * x);
  return 1.f - 2.f * __builtin_amdgcn_rcpf(e + 1.f);
#endif
}

__device__ __forceinline__ v16h pack16(v8h lo, v8h hi) {
  v16h a;
#pragma unroll
  for (int i = 0; i < 8; ++i) { a[i] = lo[i]; a[8 + i] = hi[i]; }
  return a;
}

__global__ __launch_bounds__(512, 1)
void lstm_seq_kernel(const float* __restrict__ x,
                     const float* __restrict__ W_ih,
                     const float* __restrict__ W_hh,
                     const float* __restrict__ b_ih,
                     const float* __restrict__ b_hh,
                     float* __restrict__ hpart,   // [T][8][64] node-pool partials
                     float* __restrict__ hn_out,
                     float* __restrict__ cn_out)
{
  __shared__ __align__(16) _Float16 hA[2 * HA_BUF]; // ping-pong [node][augK] f16

  const int tid  = threadIdx.x;
  const int lane = tid & 31;
  const int wv   = tid >> 5;   // 0..15
  const int mt   = wv & 3;     // M-tile (node rows 16mt..)
  const int s    = wv >> 2;    // hidden slice (cols 16s..16s+15 per gate)
  const int hi   = lane >> 4;  // lane half
  const int ln   = lane & 15;
  const int colh = 16 * s + ln;          // owned hidden column

  // zero both buffers once (pad cols 72..87 must stay zero forever)
  for (int i = tid; i < 2 * HA_BUF; i += 512) hA[i] = (_Float16)0.f;

  // per-lane gate biases (b_ih + b_hh), gate q columns = 64q + colh
  float bias[4];
#pragma unroll
  for (int q = 0; q < 4; ++q) {
    const int gi = q * 64 + colh;
    bias[q] = b_ih[gi] + b_hh[gi];
  }

  // Resident B fragments: WA[k][n]; k<64 -> W_hh[n][k]; 64<=k<72 -> W_ih[n][k-64]; else 0.
  // B lane layout (K x 16 tile): lane<16 holds col n, K=kbase..kbase+15; lane>=16 -> +16.
  v16h Bf[4][3];
#pragma unroll
  for (int q = 0; q < 4; ++q) {
    const int n = q * 64 + 16 * s + ln;  // gate column owned by this lane
#pragma unroll
    for (int kk = 0; kk < 3; ++kk) {
      const int kbase = kk * 32 + hi * 16;
      v16h b;
#pragma unroll
      for (int j = 0; j < 16; ++j) {
        const int k = kbase + j;
        float v = 0.f;
        if (k < 64)      v = W_hh[n * 64 + k];
        else if (k < 72) v = W_ih[n * 8 + (k - 64)];
        b[j] = (_Float16)v;
      }
      Bf[q][kk] = b;
    }
  }

  float c_st[8], h_st[8];
#pragma unroll
  for (int r = 0; r < 8; ++r) { c_st[r] = 0.f; h_st[r] = 0.f; }

  // x double-buffer (1 float / thread / step)
  float xcur = x[tid];            // x[0]
  float xnxt = x[OBS + tid];      // x[1]
  __syncthreads();                // zeros visible
  hA[(tid >> 3) * HA_STRIDE + 64 + (tid & 7)] = (_Float16)xcur;  // x[0] -> buf0
  xcur = xnxt;
  xnxt = x[2 * OBS + tid];
  __syncthreads();

  const int aRowOff = (16 * mt + ln) * HA_STRIDE + hi * 16;
  const int xOff    = (tid >> 3) * HA_STRIDE + 64 + (tid & 7);

  auto step = [&](const _Float16* __restrict__ rd, _Float16* __restrict__ wr, int t) {
    // ---- A fragments from LDS (16x32 f16, contiguous k = 16*half + s) ----
    const _Float16* ar = rd + aRowOff;
    v16h A[3];
#pragma unroll
    for (int kk = 0; kk < 2; ++kk) {
      A[kk] = pack16(*reinterpret_cast<const v8h*>(ar + kk * 32),
                     *reinterpret_cast<const v8h*>(ar + kk * 32 + 8));
    }
    {
      // kk=2: only augK rows 64..71 (x) have nonzero weights; upper half = 0
      const v8h lo = *reinterpret_cast<const v8h*>(ar + 64);
      v16h a;
#pragma unroll
      for (int i = 0; i < 8; ++i) { a[i] = lo[i]; a[8 + i] = (_Float16)0.f; }
      A[2] = a;
    }

    // ---- gates = [h|x] @ WA : 12 WMMAs per wave ----
    v8f acc[4] = {};
#pragma unroll
    for (int q = 0; q < 4; ++q) {
#pragma unroll
      for (int kk = 0; kk < 3; ++kk) {
        acc[q] = __builtin_amdgcn_wmma_f32_16x16x32_f16(
            false, A[kk], false, Bf[q][kk], (short)0, acc[q], false, false);
      }
    }

    // ---- LSTM cell update (branch-free transcendentals) ----
    float psum = 0.f;
#pragma unroll
    for (int r = 0; r < 8; ++r) {
      const float iv = fast_sig(acc[0][r] + bias[0]);
      const float fv = fast_sig(acc[1][r] + bias[1]);
      const float gv = fast_tanh(acc[2][r] + bias[2]);
      const float ov = fast_sig(acc[3][r] + bias[3]);
      const float c  = fv * c_st[r] + iv * gv;
      const float h  = ov * fast_tanh(c);
      c_st[r] = c; h_st[r] = h;
      psum += h;
      wr[(16 * mt + r + 8 * hi) * HA_STRIDE + colh] = (_Float16)h;
    }
    // node-pool partial -> global (reduced in head kernel, off critical path)
    hpart[t * 512 + (mt * 2 + hi) * 64 + colh] = psum;

    // stage x[t+1] into write buffer; prefetch x[t+3]
    if (t + 1 < T_SEQ) {
      wr[xOff] = (_Float16)xcur;
      xcur = xnxt;
      xnxt = (t + 3 < T_SEQ) ? x[(t + 3) * OBS + tid] : 0.f;
    }
    __syncthreads();   // single barrier per step (ping-pong => race-free)
  };

#pragma unroll 1
  for (int t = 0; t < T_SEQ; t += 2) {
    step(hA, hA + HA_BUF, t);
    step(hA + HA_BUF, hA, t + 1);
  }

  // ---- hn, cn [64 node, 64 hid] ----
#pragma unroll
  for (int r = 0; r < 8; ++r) {
    const int node = 16 * mt + r + 8 * hi;
    hn_out[node * 64 + colh] = h_st[r];
    cn_out[node * 64 + colh] = c_st[r];
  }
}

__global__ __launch_bounds__(128, 4)
void head_kernel(const float* __restrict__ hpart,
                 const float* __restrict__ W_gat,  const float* __restrict__ b_gat,
                 const float* __restrict__ W_mean, const float* __restrict__ b_mean,
                 const float* __restrict__ W_std,  const float* __restrict__ b_std,
                 float* __restrict__ out_mean, float* __restrict__ out_std)
{
  __shared__ __align__(16) _Float16 aS[16 * 72];        // hbar tile f16
  __shared__ __align__(16) _Float16 gS[16 * GS_STRIDE]; // relu'd g tile f16

  const int tid  = threadIdx.x;
  const int lane = tid & 31;
  const int wv   = tid >> 5;   // 0..3
  const int hi   = lane >> 4;
  const int ln   = lane & 15;
  const int blk  = blockIdx.x; // time rows 16*blk .. 16*blk+15

  // fold 8 node-pool partials -> hbar tile [16 x 64] -> f16
#pragma unroll
  for (int i = 0; i < 8; ++i) {
    const int idx = tid * 8 + i;              // 0..1023
    const int row = idx >> 6, col = idx & 63;
    const float* pp = hpart + (blk * 16 + row) * 512 + col;
    float sum = 0.f;
#pragma unroll
    for (int p8 = 0; p8 < 8; ++p8) sum += pp[p8 * 64];
    aS[row * 72 + col] = (_Float16)(sum * (1.f / 64.f));
  }
  __syncthreads();

  // A fragments for GEMM1 (K=64)
  v16h A1[2];
#pragma unroll
  for (int kk = 0; kk < 2; ++kk) {
    const _Float16* p = &aS[ln * 72 + kk * 32 + hi * 16];
    A1[kk] = pack16(*reinterpret_cast<const v8h*>(p),
                    *reinterpret_cast<const v8h*>(p + 8));
  }

  // GEMM1: g = relu(hbar @ W_gat^T + b_gat); wave wv owns N-tiles {4wv..4wv+3}
#pragma unroll
  for (int i = 0; i < 4; ++i) {
    const int nt = 4 * wv + i;
    const int n  = nt * 16 + ln;
    v8f acc = {};
#pragma unroll
    for (int kk = 0; kk < 2; ++kk) {
      const int kbase = kk * 32 + hi * 16;
      v16h b;
#pragma unroll
      for (int j = 0; j < 16; ++j) b[j] = (_Float16)W_gat[n * 64 + kbase + j];
      acc = __builtin_amdgcn_wmma_f32_16x16x32_f16(false, A1[kk], false, b,
                                                   (short)0, acc, false, false);
    }
    const float bg = b_gat[n];
#pragma unroll
    for (int r = 0; r < 8; ++r) {
      const float v = fmaxf(acc[r] + bg, 0.f);
      gS[(r + 8 * hi) * GS_STRIDE + n] = (_Float16)v;
    }
  }
  __syncthreads();

  // GEMM2: mean/std heads, K=256; wave wv owns output cols 16wv..16wv+15
  const int n2 = wv * 16 + ln;
  v8f accM = {}, accS = {};
#pragma unroll
  for (int kk = 0; kk < 8; ++kk) {
    const _Float16* p = &gS[ln * GS_STRIDE + kk * 32 + hi * 16];
    const v16h a = pack16(*reinterpret_cast<const v8h*>(p),
                          *reinterpret_cast<const v8h*>(p + 8));
    const int kbase = kk * 32 + hi * 16;
    v16h bm, bs;
#pragma unroll
    for (int j = 0; j < 16; ++j) {
      bm[j] = (_Float16)W_mean[n2 * 256 + kbase + j];
      bs[j] = (_Float16)W_std [n2 * 256 + kbase + j];
    }
    accM = __builtin_amdgcn_wmma_f32_16x16x32_f16(false, a, false, bm, (short)0, accM, false, false);
    accS = __builtin_amdgcn_wmma_f32_16x16x32_f16(false, a, false, bs, (short)0, accS, false, false);
  }

  const float bmv = b_mean[n2], bsv = b_std[n2];
#pragma unroll
  for (int r = 0; r < 8; ++r) {
    const int trow = blk * 16 + r + 8 * hi;
    out_mean[trow * 64 + n2] = accM[r] + bmv;
    const float xv = accS[r] + bsv;
    // branch-free softplus: max(x,0) + log(1+exp(-|x|))
    float sp = fmaxf(xv, 0.f) + __logf(1.f + __expf(-fabsf(xv)));
    sp = fminf(fmaxf(sp, 0.001f), 10.f);
    out_std[trow * 64 + n2] = sp;
  }
}

extern "C" void kernel_launch(void* const* d_in, const int* in_sizes, int n_in,
                              void* d_out, int out_size, void* d_ws, size_t ws_size,
                              hipStream_t stream) {
  const float* x      = (const float*)d_in[0];
  const float* W_ih   = (const float*)d_in[1];
  const float* W_hh   = (const float*)d_in[2];
  const float* b_ih   = (const float*)d_in[3];
  const float* b_hh   = (const float*)d_in[4];
  const float* W_gat  = (const float*)d_in[5];
  // d_in[6] (att_src), d_in[7] (att_dst): mathematically dead (identity adjacency
  // -> single-entry softmax == 1 regardless of attention logits).
  const float* b_gat  = (const float*)d_in[8];
  const float* W_mean = (const float*)d_in[9];
  const float* b_mean = (const float*)d_in[10];
  const float* W_std  = (const float*)d_in[11];
  const float* b_std  = (const float*)d_in[12];

  float* out      = (float*)d_out;
  float* out_mean = out;                    // [4096,64]
  float* out_std  = out + T_SEQ * 64;       // [4096,64]
  float* hn_out   = out + 2 * T_SEQ * 64;   // [64,64]
  float* cn_out   = hn_out + 64 * 64;       // [64,64]

  float* hpart = (float*)d_ws;              // [4096][8][64] pool partials (8 MB)

  lstm_seq_kernel<<<1, 512, 0, stream>>>(x, W_ih, W_hh, b_ih, b_hh,
                                         hpart, hn_out, cn_out);
  head_kernel<<<T_SEQ / 16, 128, 0, stream>>>(hpart, W_gat, b_gat,
                                              W_mean, b_mean, W_std, b_std,
                                              out_mean, out_std);
}
